// Net_62216896249862
// MI455X (gfx1250) — matrix operand order repaired and checked by reading, not compiled, not run on hardware
//
#include <hip/hip_runtime.h>
#include <math.h>
#include <stdint.h>

typedef __attribute__((ext_vector_type(16))) _Float16 v16h;
typedef __attribute__((ext_vector_type(8)))  _Float16 v8h;
typedef __attribute__((ext_vector_type(8)))  float    v8f;
typedef __attribute__((ext_vector_type(4)))  unsigned int v4u;
typedef __attribute__((ext_vector_type(8)))  int v8i;
typedef __attribute__((ext_vector_type(4)))  int v4i;

#define BATCH 64

#if defined(__has_builtin)
#if __has_builtin(__builtin_amdgcn_tensor_load_to_lds) && __has_builtin(__builtin_amdgcn_s_wait_tensorcnt)
#define USE_TDM 1
#endif
#endif
#ifndef USE_TDM
#define USE_TDM 0
#endif

// ---------------------------------------------------------------------------
// WMMA fragment loaders (wave32, v_wmma_f32_16x16x32_f16)
// A: 16x32 (MxK) row-major source; lane<16 holds K {0..7,16..23}, lane>=16
//    holds K {8..15,24..31} of row M=lane%16 (CDNA5 ISA A-matrix layout).
// B: 32x16 (KxN); source stored K-contiguous per column (row n of a
//    row-major [N x K] array); lane holds column N=lane%16, same K split as A.
// ---------------------------------------------------------------------------
__device__ __forceinline__ v16h frag_a_rowmajor(const _Float16* base, int ldk, int lane) {
  const int row = lane & 15;
  const int k0  = (lane & 16) ? 8 : 0;
  const _Float16* q = base + row * ldk + k0;
  v8h lo = *(const v8h*)(q);
  v8h hi = *(const v8h*)(q + 16);
  v16h r;
#pragma unroll
  for (int i = 0; i < 8; ++i) { r[i] = lo[i]; r[i + 8] = hi[i]; }
  return r;
}

__device__ __forceinline__ v16h frag_b_kcontig(const _Float16* base, int ldk, int lane) {
  const int col = lane & 15;
  const int k0  = (lane & 16) ? 8 : 0;
  const _Float16* q = base + col * ldk + k0;
  v8h lo = *(const v8h*)(q);
  v8h hi = *(const v8h*)(q + 16);
  v16h r;
#pragma unroll
  for (int i = 0; i < 8; ++i) { r[i] = lo[i]; r[i + 8] = hi[i]; }
  return r;
}

// ---------------------------------------------------------------------------
// TDM: async-load a 2D tile [rows x tileK halves] (row stride ldkHalves) from
// global into LDS at ldsAddr, row-major packed. D# built per CDNA5 ISA ch.8.
// ---------------------------------------------------------------------------
#if USE_TDM
__device__ __forceinline__ void tdm_load_tile(unsigned ldsAddr, const _Float16* gsrc,
                                              int tileK, int rows, int ldkHalves) {
  unsigned long long ga = (unsigned long long)(uintptr_t)gsrc;
  v4u g0;
  g0[0] = 1u;                                           // count=1, user mode
  g0[1] = ldsAddr;                                      // lds_addr
  g0[2] = (unsigned)(ga & 0xFFFFFFFFull);               // global_addr[31:0]
  g0[3] = (unsigned)((ga >> 32) & 0x01FFFFFFull)        // global_addr[56:32]
          | (2u << 30);                                 // type=2 ("image")
  const unsigned td0 = 0x7FFFFFFFu;                     // huge tensor dims:
  const unsigned td1 = 0x7FFFFFFFu;                     // no OOB clipping
  v8i g1;
  g1[0] = (int)(1u << 16);                              // data_size=1 (2 bytes)
  g1[1] = (int)((td0 & 0xFFFFu) << 16);                 // tensor_dim0[15:0]
  g1[2] = (int)(((td0 >> 16) & 0xFFFFu) | ((td1 & 0xFFFFu) << 16));
  g1[3] = (int)(((td1 >> 16) & 0xFFFFu) | (((unsigned)tileK & 0xFFFFu) << 16)); // tile_dim0
  g1[4] = (int)((unsigned)rows & 0xFFFFu);              // tile_dim1 (tile_dim2=0)
  g1[5] = (int)(unsigned)ldkHalves;                     // tensor_dim0_stride[31:0]
  g1[6] = 0;
  g1[7] = 0;
  v4i g2 = {0, 0, 0, 0};
  v4i g3 = {0, 0, 0, 0};
#if __clang_major__ >= 23
  v8i g4 = {0, 0, 0, 0, 0, 0, 0, 0};
  __builtin_amdgcn_tensor_load_to_lds(g0, g1, g2, g3, g4, 0);
#else
  __builtin_amdgcn_tensor_load_to_lds(g0, g1, g2, g3, 0);
#endif
}
#endif

// ---------------------------------------------------------------------------
// Convert f32 conv weights (OIHW, K=Cin*9 contiguous) to f16, K padded to
// a multiple of 32 with zeros.
// ---------------------------------------------------------------------------
__global__ void cvt_w_f16(const float* __restrict__ w, _Float16* __restrict__ o,
                          int Cout, int Ktot, int Kpad) {
  int t = blockIdx.x * 256 + threadIdx.x;
  int total = Cout * Kpad;
  if (t >= total) return;
  int k = t % Kpad; int co = t / Kpad;
  o[t] = (_Float16)((k < Ktot) ? w[(size_t)co * Ktot + k] : 0.0f);
}

// ---------------------------------------------------------------------------
// 3x3 SAME conv + bias + relu, implicit GEMM via WMMA.
// Block = 128 threads (4 waves). Wave w covers out-channels
// [blockIdx.y*64 + w*16, +16) for one 16-pixel row segment.
// A tile (64 x 32 f16 weights) staged to LDS via TDM; B tile (im2col 16x32)
// built cooperatively using precomputed k->(ci,dy,dx) LDS tables.
// ---------------------------------------------------------------------------
#define KPAD_MAX 1152

__global__ __launch_bounds__(128) void conv3x3_relu_wmma(
    const float* __restrict__ in, const _Float16* __restrict__ wf16,
    const float* __restrict__ bias, float* __restrict__ out,
    int Cin, int Cout, int H, int W, int inBatchStride, int Ktot, int Kpad)
{
  __shared__ _Float16 At[64 * 32];          // weight tile [co=64][kk=32]
  __shared__ _Float16 Bt[16 * 32];          // im2col tile [n=16][kk=32]
  __shared__ int         tOff[KPAD_MAX];    // ci*H*W + dy*W + dx
  __shared__ signed char tDy[KPAD_MAX];
  __shared__ signed char tDx[KPAD_MAX];

  const int tid  = threadIdx.x;
  const int lane = tid & 31;
  const int wave = tid >> 5;
  const int tilesX = W >> 4;
  int gx = blockIdx.x;
  const int x0 = (gx % tilesX) << 4;  gx /= tilesX;
  const int y  = gx % H;
  const int b  = gx / H;
  const int coTile0 = blockIdx.y << 6;
  const int coBase  = coTile0 + (wave << 4);
  const float* inB = in + (size_t)b * inBatchStride;

  // one-time k -> (ci,dy,dx) tables (pad region gets dy=127 sentinel -> zero)
  for (int k = tid; k < Kpad; k += 128) {
    int ci = k / 9;
    int rr = k - ci * 9;
    int dy = rr / 3 - 1;
    int dx = rr - (rr / 3) * 3 - 1;
    tOff[k] = ci * H * W + dy * W + dx;
    tDy[k]  = (signed char)((k < Ktot) ? dy : 127);
    tDx[k]  = (signed char)dx;
  }
  __syncthreads();

  const int bcol = y * W + x0;
  v8f acc = {};
  const int nChunks = Kpad >> 5;
  for (int kc = 0; kc < nChunks; ++kc) {
    const int kbase = kc << 5;
    // ---- stage A (weights) tile into LDS ----
#if USE_TDM
    if (wave == 0) {
      tdm_load_tile((unsigned)(uintptr_t)&At[0],
                    wf16 + (size_t)coTile0 * Kpad + kbase, 32, 64, Kpad);
      __builtin_amdgcn_s_wait_tensorcnt((short)0);
    }
#else
#pragma unroll
    for (int rep = 0; rep < 16; ++rep) {
      int idx = tid + rep * 128;              // 2048 halves / 128 threads
      At[idx] = wf16[(size_t)(coTile0 + (idx >> 5)) * Kpad + kbase + (idx & 31)];
    }
#endif
    // ---- build im2col B tile ----
#pragma unroll
    for (int rep = 0; rep < 4; ++rep) {
      const int idx = tid + rep * 128;        // 512 elements / 128 threads
      const int n  = idx >> 5;
      const int kk = idx & 31;
      const int k  = kbase + kk;
      const int yy = y + (int)tDy[k];
      const int xx = x0 + n + (int)tDx[k];
      float v = 0.0f;
      if ((unsigned)yy < (unsigned)H && (unsigned)xx < (unsigned)W)
        v = inB[tOff[k] + bcol + n];
      Bt[n * 32 + kk] = (_Float16)v;
    }
    __syncthreads();
    v16h afrag = frag_a_rowmajor(&At[(wave << 4) * 32], 32, lane);
    v16h bfrag = frag_b_kcontig(&Bt[0], 32, lane);
    acc = __builtin_amdgcn_wmma_f32_16x16x32_f16(false, afrag, false, bfrag,
                                                 (short)0, acc, false, false);
    __syncthreads();
  }
  // C/D layout: VGPR r, lane l -> M = r + 8*(l/16), N = l%16
  const int n  = lane & 15;
  const int mo = (lane & 16) ? 8 : 0;
  const int x  = x0 + n;
#pragma unroll
  for (int r = 0; r < 8; ++r) {
    const int co = coBase + mo + r;
    float v = acc[r] + bias[co];
    out[(((size_t)b * Cout + co) * H + y) * W + x] = v > 0.0f ? v : 0.0f;
  }
}

// ---------------------------------------------------------------------------
// 2x2 maxpool (stride 2), NCHW
// ---------------------------------------------------------------------------
__global__ void maxpool2x2(const float* __restrict__ in, float* __restrict__ out,
                           int C, int H, int W) {
  const int OH = H >> 1, OW = W >> 1;
  const int total = BATCH * C * OH * OW;
  int t = blockIdx.x * 256 + threadIdx.x;
  if (t >= total) return;
  int ox = t % OW; int r = t / OW;
  int oy = r % OH; r /= OH;
  int c  = r % C;  int b = r / C;
  const float* ib = in + (((size_t)b * C + c) * H + 2 * oy) * W + 2 * ox;
  out[t] = fmaxf(fmaxf(ib[0], ib[1]), fmaxf(ib[W], ib[W + 1]));
}

// ---------------------------------------------------------------------------
// L2 normalize over channel dim, NCHW
// ---------------------------------------------------------------------------
__global__ void l2norm_ch(const float* __restrict__ in, float* __restrict__ out,
                          int C, int HW) {
  const int total = BATCH * HW;
  int t = blockIdx.x * 256 + threadIdx.x;
  if (t >= total) return;
  int p = t % HW; int b = t / HW;
  const float* ib = in + (size_t)b * C * HW + p;
  float ss = 0.0f;
  for (int c = 0; c < C; ++c) { float v = ib[(size_t)c * HW]; ss += v * v; }
  float nrm = sqrtf(ss);
  float inv = (nrm == 0.0f) ? 1.0f : 1.0f / nrm;
  float* ob = out + (size_t)b * C * HW + p;
  for (int c = 0; c < C; ++c) ob[(size_t)c * HW] = ib[(size_t)c * HW] * inv;
}

// ---------------------------------------------------------------------------
// Extract SRxSR patches -> f16 rows P[b][i*J+j][c*4 + sy*2 + sx]
// ---------------------------------------------------------------------------
__global__ void extract_patches_f16(const float* __restrict__ f, _Float16* __restrict__ P,
                                    int C, int I, int J, int total) {
  int t = blockIdx.x * 256 + threadIdx.x;
  if (t >= total) return;
  const int d = C * 4;
  const int R = I * J;
  int dc  = t % d;   int rest = t / d;
  int rij = rest % R; int b = rest / R;
  int c = dc >> 2; int sy = (dc >> 1) & 1; int sx = dc & 1;
  int i = rij / J, j = rij % J;
  const int H = 2 * I, W = 2 * J;
  P[t] = (_Float16)f[(((size_t)b * C + c) * H + (2 * i + sy)) * W + (2 * j + sx)];
}

// ---------------------------------------------------------------------------
// Correlation GEMM: S[b] = P1[b] (R x d) * P2[b]^T -> (R x R), f32 accum.
// Block = 128 (4 waves); wave w does N-tile blockIdx.y*4 + w.
// ---------------------------------------------------------------------------
__global__ __launch_bounds__(128) void corr_gemm_wmma(
    const _Float16* __restrict__ P1, const _Float16* __restrict__ P2,
    float* __restrict__ S, int R, int d)
{
  const int lane = threadIdx.x & 31;
  const int wave = threadIdx.x >> 5;
  const int mt = blockIdx.x;
  const int nt = (blockIdx.y << 2) + wave;
  const int b  = blockIdx.z;
  const _Float16* Abase = P1 + ((size_t)b * R + mt * 16) * d;
  const _Float16* Bbase = P2 + ((size_t)b * R + nt * 16) * d;
  v8f acc = {};
  for (int kc = 0; kc < d; kc += 32) {
    __builtin_prefetch(Abase + kc + 32, 0, 3);
    __builtin_prefetch(Bbase + kc + 32, 0, 3);
    v16h afrag = frag_a_rowmajor(Abase + kc, d, lane);
    v16h bfrag = frag_b_kcontig(Bbase + kc, d, lane);
    acc = __builtin_amdgcn_wmma_f32_16x16x32_f16(false, afrag, false, bfrag,
                                                 (short)0, acc, false, false);
  }
  const int n  = nt * 16 + (lane & 15);
  const int m0 = mt * 16 + ((lane & 16) ? 8 : 0);
  float* Sb = S + (size_t)b * R * R;
#pragma unroll
  for (int r = 0; r < 8; ++r)
    Sb[(size_t)(m0 + r) * R + n] = acc[r];
}

// ---------------------------------------------------------------------------
// Row softmax + expectation -> cor[b][0][i][j]=E[row]-i, cor[b][1][i][j]=E[col]-j
// One block (256 threads) per (b, row).
// ---------------------------------------------------------------------------
__global__ __launch_bounds__(256) void softmax_expect(
    const float* __restrict__ S, float* __restrict__ cor, int I, int J)
{
  const int R = I * J;
  const int row = blockIdx.x % R;
  const int b   = blockIdx.x / R;
  const float* s = S + ((size_t)b * R + row) * R;
  __shared__ float r0[256], r1[256], r2[256];
  const int tx = threadIdx.x;

  float m = -3.4e38f;
  for (int k = tx; k < R; k += 256) m = fmaxf(m, s[k]);
  r0[tx] = m; __syncthreads();
  for (int st = 128; st > 0; st >>= 1) {
    if (tx < st) r0[tx] = fmaxf(r0[tx], r0[tx + st]);
    __syncthreads();
  }
  m = r0[0];
  __syncthreads();

  float sum = 0.0f, er = 0.0f, ec = 0.0f;
  for (int k = tx; k < R; k += 256) {
    float t = __expf(s[k] - m);
    sum += t;
    er  += t * (float)(k / J);
    ec  += t * (float)(k % J);
  }
  r0[tx] = sum; r1[tx] = er; r2[tx] = ec; __syncthreads();
  for (int st = 128; st > 0; st >>= 1) {
    if (tx < st) { r0[tx] += r0[tx + st]; r1[tx] += r1[tx + st]; r2[tx] += r2[tx + st]; }
    __syncthreads();
  }
  if (tx == 0) {
    float inv = 1.0f / r0[0];
    int i = row / J, j = row % J;
    cor[(((size_t)b * 2 + 0) * I + i) * J + j] = r1[0] * inv - (float)i;
    cor[(((size_t)b * 2 + 1) * I + i) * J + j] = r2[0] * inv - (float)j;
  }
}

// ---------------------------------------------------------------------------
// MLP: h = relu(x @ w1 + b1) ; off = (h @ w2 + b2) * SR
// ---------------------------------------------------------------------------
__global__ void mlp1(const float* __restrict__ x, const float* __restrict__ w1,
                     const float* __restrict__ b1, float* __restrict__ h, int K) {
  int t = blockIdx.x * 256 + threadIdx.x;
  if (t >= BATCH * 512) return;
  int j = t & 511; int b = t >> 9;
  const float* xb = x + (size_t)b * K;
  float acc = b1[j];
  for (int k = 0; k < K; ++k) acc += xb[k] * w1[(size_t)k * 512 + j];
  h[t] = acc > 0.0f ? acc : 0.0f;
}

__global__ void mlp2(const float* __restrict__ h, const float* __restrict__ w2,
                     const float* __restrict__ b2, float* __restrict__ off,
                     float* __restrict__ outslot) {
  int t = blockIdx.x * 256 + threadIdx.x;
  if (t >= BATCH * 8) return;
  int j = t & 7; int b = t >> 3;
  const float* hb = h + (size_t)b * 512;
  float acc = b2[j];
  for (int k = 0; k < 512; ++k) acc += hb[k] * w2[k * 8 + j];
  acc *= 2.0f;  // SR
  off[t] = acc;
  outslot[t] = acc;
}

__global__ void vadd(const float* __restrict__ a, const float* __restrict__ b,
                     float* __restrict__ c, int n) {
  int t = blockIdx.x * 256 + threadIdx.x;
  if (t < n) c[t] = a[t] + b[t];
}

// ---------------------------------------------------------------------------
// Per-batch DLT (8x8 Gaussian elimination w/ partial pivot) + norm_H
// ---------------------------------------------------------------------------
__global__ __launch_bounds__(64) void dlt_norm(const float* __restrict__ off,
                                               float scale, float ps,
                                               float* __restrict__ Hout) {
  int b = threadIdx.x;
  if (b >= BATCH) return;
  const float sx[4] = {0.0f, ps - 1.0f, 0.0f, ps - 1.0f};
  const float sy[4] = {0.0f, 0.0f, ps - 1.0f, ps - 1.0f};
  float u[4], v[4];
  for (int p = 0; p < 4; ++p) {
    u[p] = sx[p] + off[b * 8 + 2 * p + 0] * scale;
    v[p] = sy[p] + off[b * 8 + 2 * p + 1] * scale;
  }
  float A[8][9];
  for (int p = 0; p < 4; ++p) {
    float x = sx[p], y = sy[p];
    float* ra = A[p];
    ra[0] = x; ra[1] = y; ra[2] = 1; ra[3] = 0; ra[4] = 0; ra[5] = 0;
    ra[6] = -u[p] * x; ra[7] = -u[p] * y; ra[8] = u[p];
    float* rb = A[p + 4];
    rb[0] = 0; rb[1] = 0; rb[2] = 0; rb[3] = x; rb[4] = y; rb[5] = 1;
    rb[6] = -v[p] * x; rb[7] = -v[p] * y; rb[8] = v[p];
  }
  for (int c = 0; c < 8; ++c) {
    int piv = c; float best = fabsf(A[c][c]);
    for (int r = c + 1; r < 8; ++r) {
      float av = fabsf(A[r][c]);
      if (av > best) { best = av; piv = r; }
    }
    if (piv != c)
      for (int k = c; k < 9; ++k) { float tmp = A[c][k]; A[c][k] = A[piv][k]; A[piv][k] = tmp; }
    float inv = 1.0f / A[c][c];
    for (int r = c + 1; r < 8; ++r) {
      float f = A[r][c] * inv;
      for (int k = c; k < 9; ++k) A[r][k] -= f * A[c][k];
    }
  }
  float hs[8];
  for (int r = 7; r >= 0; --r) {
    float s = A[r][8];
    for (int k = r + 1; k < 8; ++k) s -= A[r][k] * hs[k];
    hs[r] = s / A[r][r];
  }
  float Hm[9] = {hs[0], hs[1], hs[2], hs[3], hs[4], hs[5], hs[6], hs[7], 1.0f};
  const float hp = ps * 0.5f, ihp = 1.0f / hp;
  float T[9];
  for (int r = 0; r < 3; ++r) {  // T = Hm * M
    float h0 = Hm[r * 3], h1 = Hm[r * 3 + 1], h2 = Hm[r * 3 + 2];
    T[r * 3 + 0] = hp * h0;
    T[r * 3 + 1] = hp * h1;
    T[r * 3 + 2] = hp * h0 + hp * h1 + h2;
  }
  for (int c = 0; c < 3; ++c) {  // Hn = Minv * T
    Hout[b * 9 + 0 * 3 + c] = ihp * T[0 * 3 + c] - T[2 * 3 + c];
    Hout[b * 9 + 1 * 3 + c] = ihp * T[1 * 3 + c] - T[2 * 3 + c];
    Hout[b * 9 + 2 * 3 + c] = T[2 * 3 + c];
  }
}

// ---------------------------------------------------------------------------
// Homography bilinear warp: out[b][c] = sample(feat[b][c], H[b])
// ---------------------------------------------------------------------------
__global__ void warp_bilinear(const float* __restrict__ feat, const float* __restrict__ Hn,
                              float* __restrict__ out, int C, int HH, int WW,
                              int OH, int OW) {
  const int total = BATCH * OH * OW;
  int t = blockIdx.x * 256 + threadIdx.x;
  if (t >= total) return;
  int ox = t % OW; int rest = t / OW;
  int oy = rest % OH; int b = rest / OH;
  const float* Hb = Hn + b * 9;
  float gx = -1.0f + 2.0f * (float)ox / (float)(OW - 1);
  float gy = -1.0f + 2.0f * (float)oy / (float)(OH - 1);
  float X = Hb[0] * gx + Hb[1] * gy + Hb[2];
  float Y = Hb[3] * gx + Hb[4] * gy + Hb[5];
  float Z = Hb[6] * gx + Hb[7] * gy + Hb[8];
  if (fabsf(Z) < 1e-7f) Z = 1e-7f;
  float xs = X / Z, ys = Y / Z;
  float xi = (xs + 1.0f) * (float)(WW - 1) * 0.5f;
  float yi = (ys + 1.0f) * (float)(HH - 1) * 0.5f;
  float x0f = floorf(xi), y0f = floorf(yi);
  int x0 = min(max((int)x0f, 0), WW - 1);
  int x1 = min(max((int)x0f + 1, 0), WW - 1);
  int y0 = min(max((int)y0f, 0), HH - 1);
  int y1 = min(max((int)y0f + 1, 0), HH - 1);
  float wa = (x0f + 1.0f - xi) * (y0f + 1.0f - yi);
  float wb = (x0f + 1.0f - xi) * (yi - y0f);
  float wc = (xi - x0f) * (y0f + 1.0f - yi);
  float wd = (xi - x0f) * (yi - y0f);
  for (int c = 0; c < C; ++c) {
    const float* fb = feat + ((size_t)b * C + c) * HH * WW;
    float Ia = fb[y0 * WW + x0], Ib = fb[y1 * WW + x0];
    float Ic = fb[y0 * WW + x1], Id = fb[y1 * WW + x1];
    out[((size_t)b * C + c) * OH * OW + oy * OW + ox] =
        wa * Ia + wb * Ib + wc * Ic + wd * Id;
  }
}

// ===========================================================================
extern "C" void kernel_launch(void* const* d_in, const int* in_sizes, int n_in,
                              void* d_out, int out_size, void* d_ws, size_t ws_size,
                              hipStream_t stream) {
  (void)in_sizes; (void)n_in; (void)out_size; (void)ws_size;
  const float* img = (const float*)d_in[0];
  const float* cw[8]; const float* cb[8];
  for (int i = 0; i < 8; ++i) { cw[i] = (const float*)d_in[1 + 2 * i]; cb[i] = (const float*)d_in[2 + 2 * i]; }
  const float* rw1[3] = {(const float*)d_in[17], (const float*)d_in[21], (const float*)d_in[25]};
  const float* rb1[3] = {(const float*)d_in[18], (const float*)d_in[22], (const float*)d_in[26]};
  const float* rw2[3] = {(const float*)d_in[19], (const float*)d_in[23], (const float*)d_in[27]};
  const float* rb2[3] = {(const float*)d_in[20], (const float*)d_in[24], (const float*)d_in[28]};
  float* outp = (float*)d_out;

  // ---- bump allocator over d_ws ----
  char* wsb = (char*)d_ws;
  size_t cur = 0;
  auto carve = [&](size_t bytes) -> char* {
    char* p = wsb + cur;
    cur = (cur + bytes + 255) & ~(size_t)255;
    return p;
  };

  // conv layer geometry
  const int CI[8] = {1, 64, 64, 64, 64, 128, 128, 128};
  const int CO[8] = {64, 64, 64, 64, 128, 128, 128, 128};
  int KT[8], KP[8];
  for (int i = 0; i < 8; ++i) { KT[i] = CI[i] * 9; KP[i] = ((KT[i] + 31) / 32) * 32; }

  _Float16* wf[8];
  for (int i = 0; i < 8; ++i) wf[i] = (_Float16*)carve((size_t)CO[i] * KP[i] * 2);

  float* bufA = (float*)carve((size_t)67108864 * 4);
  float* bufB = (float*)carve((size_t)67108864 * 4);
  float* f1[2], *f2[2], *f3[2];
  for (int i = 0; i < 2; ++i) f1[i] = (float*)carve((size_t)16777216 * 4);
  for (int i = 0; i < 2; ++i) f2[i] = (float*)carve((size_t)8388608 * 4);
  for (int i = 0; i < 2; ++i) f3[i] = (float*)carve((size_t)2097152 * 4);
  _Float16* Pa = (_Float16*)carve((size_t)16777216 * 2);
  _Float16* Pb = (_Float16*)carve((size_t)16777216 * 2);
  float* S    = (float*)carve((size_t)67108864 * 4);
  float* cor  = (float*)carve((size_t)131072 * 4);
  float* hbuf = (float*)carve((size_t)BATCH * 512 * 4);
  float* off1   = (float*)carve(512 * 4);
  float* off2   = (float*)carve(512 * 4);
  float* off3   = (float*)carve(512 * 4);
  float* off12  = (float*)carve(512 * 4);
  float* off123 = (float*)carve(512 * 4);
  float* H1n = (float*)carve(576 * 4);
  float* H2n = (float*)carve(576 * 4);
  // reuse the big conv ping-pong buffers for post-feature stages
  float* L1a = bufA;               float* L1b = bufA + 16777216;  float* W1 = bufA + 2 * 16777216;
  float* L2a = bufB;               float* L2b = bufB + 8388608;   float* W2 = bufB + 2 * 8388608;
  float* L3a = bufB + 3 * 8388608; float* L3b = L3a + 2097152;

  // ---- convert conv weights to f16 (K-padded) ----
  for (int i = 0; i < 8; ++i) {
    int total = CO[i] * KP[i];
    cvt_w_f16<<<(total + 255) / 256, 256, 0, stream>>>(cw[i], wf[i], CO[i], KT[i], KP[i]);
  }

  // ---- feature extraction for both images ----
  for (int im = 0; im < 2; ++im) {
    const float* in0 = img + (size_t)im * 128 * 128;
    conv3x3_relu_wmma<<<dim3(64 * 128 * 8, 1), 128, 0, stream>>>(
        in0, wf[0], cb[0], bufA, 1, 64, 128, 128, 2 * 128 * 128, KT[0], KP[0]);
    conv3x3_relu_wmma<<<dim3(64 * 128 * 8, 1), 128, 0, stream>>>(
        bufA, wf[1], cb[1], bufB, 64, 64, 128, 128, 64 * 128 * 128, KT[1], KP[1]);
    maxpool2x2<<<(64 * 64 * 64 * 64 + 255) / 256, 256, 0, stream>>>(bufB, bufA, 64, 128, 128);
    conv3x3_relu_wmma<<<dim3(64 * 64 * 4, 1), 128, 0, stream>>>(
        bufA, wf[2], cb[2], bufB, 64, 64, 64, 64, 64 * 64 * 64, KT[2], KP[2]);
    conv3x3_relu_wmma<<<dim3(64 * 64 * 4, 1), 128, 0, stream>>>(
        bufB, wf[3], cb[3], f1[im], 64, 64, 64, 64, 64 * 64 * 64, KT[3], KP[3]);
    maxpool2x2<<<(64 * 64 * 32 * 32 + 255) / 256, 256, 0, stream>>>(f1[im], bufA, 64, 64, 64);
    conv3x3_relu_wmma<<<dim3(64 * 32 * 2, 2), 128, 0, stream>>>(
        bufA, wf[4], cb[4], bufB, 64, 128, 32, 32, 64 * 32 * 32, KT[4], KP[4]);
    conv3x3_relu_wmma<<<dim3(64 * 32 * 2, 2), 128, 0, stream>>>(
        bufB, wf[5], cb[5], f2[im], 128, 128, 32, 32, 128 * 32 * 32, KT[5], KP[5]);
    maxpool2x2<<<(64 * 128 * 16 * 16 + 255) / 256, 256, 0, stream>>>(f2[im], bufA, 128, 32, 32);
    conv3x3_relu_wmma<<<dim3(64 * 16 * 1, 2), 128, 0, stream>>>(
        bufA, wf[6], cb[6], bufB, 128, 128, 16, 16, 128 * 16 * 16, KT[6], KP[6]);
    conv3x3_relu_wmma<<<dim3(64 * 16 * 1, 2), 128, 0, stream>>>(
        bufB, wf[7], cb[7], f3[im], 128, 128, 16, 16, 128 * 16 * 16, KT[7], KP[7]);
  }

  // ---- scale 1: correlation on f3 (128ch, 16x16 -> I=J=8, R=64, d=512) ----
  l2norm_ch<<<(BATCH * 256 + 255) / 256, 256, 0, stream>>>(f3[0], L3a, 128, 256);
  l2norm_ch<<<(BATCH * 256 + 255) / 256, 256, 0, stream>>>(f3[1], L3b, 128, 256);
  extract_patches_f16<<<(2097152 + 255) / 256, 256, 0, stream>>>(L3a, Pa, 128, 8, 8, 2097152);
  extract_patches_f16<<<(2097152 + 255) / 256, 256, 0, stream>>>(L3b, Pb, 128, 8, 8, 2097152);
  corr_gemm_wmma<<<dim3(4, 1, BATCH), 128, 0, stream>>>(Pa, Pb, S, 64, 512);
  softmax_expect<<<BATCH * 64, 256, 0, stream>>>(S, cor, 8, 8);
  mlp1<<<(BATCH * 512 + 255) / 256, 256, 0, stream>>>(cor, rw1[0], rb1[0], hbuf, 128);
  mlp2<<<(BATCH * 8 + 255) / 256, 256, 0, stream>>>(hbuf, rw2[0], rb2[0], off1, outp + 1728);
  dlt_norm<<<1, 64, 0, stream>>>(off1, 0.25f, 32.0f, H1n);

  // ---- scale 2: warp l2n(f22), correlate with l2n(f12) (I=J=16, R=256, d=512) ----
  l2norm_ch<<<(BATCH * 1024 + 255) / 256, 256, 0, stream>>>(f2[1], L2b, 128, 1024);
  warp_bilinear<<<(BATCH * 32 * 32 + 255) / 256, 256, 0, stream>>>(L2b, H1n, W2, 128, 32, 32, 32, 32);
  l2norm_ch<<<(BATCH * 1024 + 255) / 256, 256, 0, stream>>>(f2[0], L2a, 128, 1024);
  extract_patches_f16<<<(8388608 + 255) / 256, 256, 0, stream>>>(L2a, Pa, 128, 16, 16, 8388608);
  extract_patches_f16<<<(8388608 + 255) / 256, 256, 0, stream>>>(W2, Pb, 128, 16, 16, 8388608);
  corr_gemm_wmma<<<dim3(16, 4, BATCH), 128, 0, stream>>>(Pa, Pb, S, 256, 512);
  softmax_expect<<<BATCH * 256, 256, 0, stream>>>(S, cor, 16, 16);
  mlp1<<<(BATCH * 512 + 255) / 256, 256, 0, stream>>>(cor, rw1[1], rb1[1], hbuf, 512);
  mlp2<<<(BATCH * 8 + 255) / 256, 256, 0, stream>>>(hbuf, rw2[1], rb2[1], off2, outp + 2240);
  vadd<<<2, 256, 0, stream>>>(off1, off2, off12, 512);
  dlt_norm<<<1, 64, 0, stream>>>(off12, 0.5f, 64.0f, H2n);

  // ---- scale 3: warp l2n(f21), correlate with l2n(f11) (I=J=32, R=1024, d=256) ----
  l2norm_ch<<<(BATCH * 4096 + 255) / 256, 256, 0, stream>>>(f1[1], L1b, 64, 4096);
  warp_bilinear<<<(BATCH * 64 * 64 + 255) / 256, 256, 0, stream>>>(L1b, H2n, W1, 64, 64, 64, 64, 64);
  l2norm_ch<<<(BATCH * 4096 + 255) / 256, 256, 0, stream>>>(f1[0], L1a, 64, 4096);
  extract_patches_f16<<<(16777216 + 255) / 256, 256, 0, stream>>>(L1a, Pa, 64, 32, 32, 16777216);
  extract_patches_f16<<<(16777216 + 255) / 256, 256, 0, stream>>>(W1, Pb, 64, 32, 32, 16777216);
  corr_gemm_wmma<<<dim3(64, 16, BATCH), 128, 0, stream>>>(Pa, Pb, S, 1024, 256);
  softmax_expect<<<BATCH * 1024, 256, 0, stream>>>(S, cor, 32, 32);
  mlp1<<<(BATCH * 512 + 255) / 256, 256, 0, stream>>>(cor, rw1[2], rb1[2], hbuf, 2048);
  mlp2<<<(BATCH * 8 + 255) / 256, 256, 0, stream>>>(hbuf, rw2[2], rb2[2], off3, outp + 2752);
  vadd<<<2, 256, 0, stream>>>(off12, off3, off123, 512);

  // ---- final homographies at ps=128 ----
  dlt_norm<<<1, 64, 0, stream>>>(off1,   1.0f, 128.0f, outp + 0);
  dlt_norm<<<1, 64, 0, stream>>>(off12,  1.0f, 128.0f, outp + 576);
  dlt_norm<<<1, 64, 0, stream>>>(off123, 1.0f, 128.0f, outp + 1152);
}
